// Block_61744449848022
// MI455X (gfx1250) — matrix operand order repaired and checked by reading, not compiled, not run on hardware
//
#include <hip/hip_runtime.h>

typedef __attribute__((ext_vector_type(16))) __bf16 v16bf;
typedef __attribute__((ext_vector_type(8)))  __bf16 v8bf;
typedef __attribute__((ext_vector_type(8)))  float  v8f;

#define SEQ    2048
#define HIDDEN 1024
#define HEADS  16
#define HDIM   64
#define QKVP   7168
#define VP     5120
#define EPSV   1e-5f
#define NEGV   -1e10f

static __device__ __forceinline__ v16bf mk16(v8bf lo, v8bf hi) {
  return __builtin_shufflevector(lo, hi, 0,1,2,3,4,5,6,7,8,9,10,11,12,13,14,15);
}

// gfx1250 async copy: 16B global -> LDS, tracked by ASYNCcnt (no VGPR staging)
static __device__ __forceinline__ void async_copy_b128(unsigned lds_off, const void* gptr) {
  asm volatile("global_load_async_to_lds_b128 %0, %1, off"
               :: "v"(lds_off), "v"((unsigned long long)(size_t)gptr)
               : "memory");
}
static __device__ __forceinline__ void wait_async0() {
  asm volatile("s_wait_asynccnt 0x0" ::: "memory");
}

// ---------------------------------------------------------------- fp32->bf16
__global__ __launch_bounds__(256) void f2bf_kernel(const float* __restrict__ in,
                                                   __bf16* __restrict__ out, int n) {
  int i = blockIdx.x * 256 + threadIdx.x;
  if (i < n) out[i] = (__bf16)in[i];
}

// ---------------------------------------------- in_ln + token shift -> bf16 h
__global__ __launch_bounds__(256) void ln_shift_kernel(const float* __restrict__ x,
                                                       const float* __restrict__ g,
                                                       const float* __restrict__ b,
                                                       __bf16* __restrict__ h) {
  __shared__ float s1[256], s2[256];
  const int l = blockIdx.x, tid = threadIdx.x;
  const float* xr = x + (size_t)l * HIDDEN;
  float sum = 0.f, sq = 0.f;
#pragma unroll
  for (int i = 0; i < 4; i++) { float v = xr[tid + i * 256]; sum += v; sq += v * v; }
  s1[tid] = sum; s2[tid] = sq; __syncthreads();
  for (int off = 128; off > 0; off >>= 1) {
    if (tid < off) { s1[tid] += s1[tid + off]; s2[tid] += s2[tid + off]; }
    __syncthreads();
  }
  float mean = s1[0] * (1.f / HIDDEN);
  float var  = s2[0] * (1.f / HIDDEN) - mean * mean;
  float rinv = rsqrtf(var + EPSV);
#pragma unroll
  for (int i = 0; i < 4; i++) {
    int c = tid + i * 256;
    float y = (xr[c] - mean) * rinv * g[c] + b[c];
    if (c >= 256)            h[(size_t)l * HIDDEN + c]       = (__bf16)y;   // unshifted channels
    else if (l + 1 < SEQ)    h[(size_t)(l + 1) * HIDDEN + c] = (__bf16)y;   // shifted fwd 1 token
  }
  if (l == 0) h[tid] = (__bf16)0.0f;  // zero-pad front, channels 0..255
}

// --------------------------------------------------------------- NT GEMM bf16
// C[M,N] = alpha * A[M,K] * W[N,K]^T (row-major). Block tile 64x256x32,
// 8 waves, each wave 2(M) x 4(N) = 8 WMMAs per k-step. Tiles staged to LDS
// via gfx1250 global_load_async_to_lds_b128 (ASYNCcnt).
__global__ __launch_bounds__(256) void gemm_nt_bf16(const __bf16* __restrict__ A,
                                                    const __bf16* __restrict__ W,
                                                    float* __restrict__ C,
                                                    int M, int N, int K, float alpha) {
  __shared__ __bf16 As[64][32];    //  4 KB
  __shared__ __bf16 Ws[256][32];   // 16 KB
  const int tid  = threadIdx.x;
  const int n0   = blockIdx.x * 256;
  const int m0   = blockIdx.y * 64;
  const int wid  = tid >> 5, lane = tid & 31;
  const int half = lane >> 4, mr = lane & 15;
  const int wm   = wid >> 2, wn = wid & 3;   // 2 x 4 wave grid

  v8f acc[2][4] = {};

  // cooperative staging: each thread 1 x b128 of A, 4 x b128 of W
  const int arow = tid >> 2, acol = (tid & 3) * 8;
  const unsigned a_lds = (unsigned)(size_t)&As[arow][acol];
  unsigned w_lds[4];
#pragma unroll
  for (int i = 0; i < 4; i++)
    w_lds[i] = (unsigned)(size_t)&Ws[arow + i * 64][acol];

  for (int kb = 0; kb < K; kb += 32) {
    __syncthreads();   // previous iteration's fragment reads complete
    async_copy_b128(a_lds, A + (size_t)(m0 + arow) * K + kb + acol);
#pragma unroll
    for (int i = 0; i < 4; i++)
      async_copy_b128(w_lds[i], W + (size_t)(n0 + arow + i * 64) * K + kb + acol);
    wait_async0();     // our lane's async copies are in LDS
    __syncthreads();   // everyone's are

    // A fragments (16x32): lane=row, k chunks {half*8..+7},{16+half*8..+7}
    v16bf a[2];
#pragma unroll
    for (int s = 0; s < 2; s++) {
      int row = wm * 32 + s * 16 + mr;
      a[s] = mk16(*(const v8bf*)&As[row][half * 8],
                  *(const v8bf*)&As[row][16 + half * 8]);
    }
    // B fragments (32x16): lane=col, k chunks {half*8..+7},{16+half*8..+7}
#pragma unroll
    for (int t = 0; t < 4; t++) {
      int row = wn * 64 + t * 16 + mr;
      v16bf bfr = mk16(*(const v8bf*)&Ws[row][half * 8],
                       *(const v8bf*)&Ws[row][16 + half * 8]);
#pragma unroll
      for (int s = 0; s < 2; s++)
        acc[s][t] = __builtin_amdgcn_wmma_f32_16x16x32_bf16(false, a[s], false, bfr,
                                                            (short)0, acc[s][t], false, false);
    }
  }
#pragma unroll
  for (int s = 0; s < 2; s++)
#pragma unroll
    for (int t = 0; t < 4; t++)
#pragma unroll
      for (int r = 0; r < 8; r++) {   // C layout: row = r + 8*half, col = lane&15
        int m = m0 + wm * 32 + s * 16 + r + half * 8;
        C[(size_t)m * N + n0 + wn * 64 + t * 16 + mr] = alpha * acc[s][t][r];
      }
}

// ----------------------------------------------- q_ln / k_ln / v -> bf16 bufs
static __device__ __forceinline__ void ln_seg(const float* __restrict__ seg,
                                              const float* __restrict__ g,
                                              const float* __restrict__ b,
                                              __bf16* __restrict__ out,
                                              float* s1, float* s2, int tid) {
  float sum = 0.f, sq = 0.f;
#pragma unroll
  for (int i = 0; i < 4; i++) { float v = seg[tid + i * 256]; sum += v; sq += v * v; }
  s1[tid] = sum; s2[tid] = sq; __syncthreads();
  for (int off = 128; off > 0; off >>= 1) {
    if (tid < off) { s1[tid] += s1[tid + off]; s2[tid] += s2[tid + off]; }
    __syncthreads();
  }
  float mean = s1[0] * (1.f / 1024.f);
  float var  = s2[0] * (1.f / 1024.f) - mean * mean;
  float rinv = rsqrtf(var + EPSV);
  __syncthreads();
#pragma unroll
  for (int i = 0; i < 4; i++) {
    int c = tid + i * 256;
    out[c] = (__bf16)((seg[c] - mean) * rinv * g[c] + b[c]);
  }
}

__global__ __launch_bounds__(256) void qkv_ln_kernel(const float* __restrict__ qkvp,
                                                     const float* __restrict__ qg, const float* __restrict__ qb,
                                                     const float* __restrict__ kg, const float* __restrict__ kb,
                                                     __bf16* __restrict__ Q, __bf16* __restrict__ Ko,
                                                     __bf16* __restrict__ V) {
  __shared__ float s1[256], s2[256];
  const int l = blockIdx.x, tid = threadIdx.x;
  const float* row = qkvp + (size_t)l * QKVP;
  ln_seg(row,           qg, qb, Q  + (size_t)l * HIDDEN, s1, s2, tid);
  ln_seg(row + HIDDEN,  kg, kb, Ko + (size_t)l * HIDDEN, s1, s2, tid);
#pragma unroll
  for (int i = 0; i < 4; i++) {
    int c = tid + i * 256;
    V[(size_t)l * HIDDEN + c] = (__bf16)row[2 * HIDDEN + c];
  }
}

// ------------------------------------------------------------ flash attention
__global__ __launch_bounds__(32) void attn_kernel(const __bf16* __restrict__ Q,
                                                  const __bf16* __restrict__ Kb,
                                                  const __bf16* __restrict__ V,
                                                  const float* __restrict__ AL,
                                                  __bf16* __restrict__ cat) {
  __shared__ float Pl[16][33];
  const int qb = blockIdx.x, h = blockIdx.y;
  const int lane = threadIdx.x, half = lane >> 4, mr = lane & 15;
  const float slope = exp2f(-0.5f * (float)(h + 1));  // 16 heads: 2^(-(i+1)/2)

  // Q A-fragments: 2 chunks of 32 over head_dim
  v16bf qa[2];
#pragma unroll
  for (int c = 0; c < 2; c++) {
    const __bf16* base = Q + (size_t)(qb * 16 + mr) * HIDDEN + h * HDIM + c * 32;
    qa[c] = mk16(*(const v8bf*)(base + half * 8), *(const v8bf*)(base + 16 + half * 8));
  }

  v8f o[4] = {};
  float mi[8], li[8];
#pragma unroll
  for (int r = 0; r < 8; r++) { mi[r] = -3e38f; li[r] = 0.f; }

  const int imax = qb * 16 + 15;
  for (int jb = 0; jb <= imax; jb += 32) {
    // S = Q K^T for two 16-wide j tiles
    v8f s0 = {}, s1v = {};
#pragma unroll
    for (int c = 0; c < 2; c++) {
      const __bf16* k0 = Kb + (size_t)(jb + mr) * HIDDEN + h * HDIM + c * 32;
      const __bf16* k1 = Kb + (size_t)(jb + 16 + mr) * HIDDEN + h * HDIM + c * 32;
      v16bf bk0 = mk16(*(const v8bf*)(k0 + half * 8), *(const v8bf*)(k0 + 16 + half * 8));
      v16bf bk1 = mk16(*(const v8bf*)(k1 + half * 8), *(const v8bf*)(k1 + 16 + half * 8));
      s0  = __builtin_amdgcn_wmma_f32_16x16x32_bf16(false, qa[c], false, bk0, (short)0, s0,  false, false);
      s1v = __builtin_amdgcn_wmma_f32_16x16x32_bf16(false, qa[c], false, bk1, (short)0, s1v, false, false);
    }
    const int j0 = jb + mr, j1 = jb + 16 + mr;
#pragma unroll
    for (int r = 0; r < 8; r++) {
      int i = qb * 16 + r + half * 8;
      float a0 = s0[r]  * 0.125f + AL[(size_t)i * SEQ + j0] + slope * (float)j0 + (j0 <= i ? 0.f : NEGV);
      float a1 = s1v[r] * 0.125f + AL[(size_t)i * SEQ + j1] + slope * (float)j1 + (j1 <= i ? 0.f : NEGV);
      float rm = fmaxf(a0, a1);                       // row reduce within 16-lane half
#pragma unroll
      for (int msk = 8; msk >= 1; msk >>= 1) rm = fmaxf(rm, __shfl_xor(rm, msk, 32));
      float mn = fmaxf(mi[r], rm);
      float sc = __expf(mi[r] - mn);
      float p0 = __expf(a0 - mn), p1 = __expf(a1 - mn);
      float rs = p0 + p1;
#pragma unroll
      for (int msk = 8; msk >= 1; msk >>= 1) rs += __shfl_xor(rs, msk, 32);
      li[r] = li[r] * sc + rs;
      mi[r] = mn;
#pragma unroll
      for (int t = 0; t < 4; t++) o[t][r] *= sc;
      Pl[r + half * 8][mr]      = p0;                 // C-layout -> plain [row][j]
      Pl[r + half * 8][16 + mr] = p1;
    }
    __syncthreads();
    // transpose P into A-fragment layout (row = lane&15, k per A map)
    v16bf pa;
#pragma unroll
    for (int e = 0; e < 16; e++) {
      int kl = (e < 8 ? e : e + 8) + half * 8;
      pa[e] = (__bf16)Pl[mr][kl];
    }
    __syncthreads();
    // O += P V  (B: lane=head-dim col, k per A map over 32 keys)
#pragma unroll
    for (int t = 0; t < 4; t++) {
      v16bf vb;
#pragma unroll
      for (int e = 0; e < 16; e++) {
        int j = jb + (e < 8 ? e : e + 8) + half * 8;
        vb[e] = V[(size_t)j * HIDDEN + h * HDIM + t * 16 + mr];
      }
      o[t] = __builtin_amdgcn_wmma_f32_16x16x32_bf16(false, pa, false, vb, (short)0, o[t], false, false);
    }
  }
#pragma unroll
  for (int r = 0; r < 8; r++) {
    int i = qb * 16 + r + half * 8;
    float rl = 1.f / li[r];
    size_t base = (size_t)i * VP + h * HDIM + mr;
#pragma unroll
    for (int t = 0; t < 4; t++) cat[base + t * 16] = (__bf16)(o[t][r] * rl);
  }
}

// ------------------------------------------------------- mid_ln + relu -> cat
__global__ __launch_bounds__(256) void mid_ln_kernel(const float* __restrict__ qkvp,
                                                     const float* __restrict__ g,
                                                     const float* __restrict__ b,
                                                     __bf16* __restrict__ cat) {
  __shared__ float s1[256], s2[256];
  const int l = blockIdx.x, tid = threadIdx.x;
  const float* seg = qkvp + (size_t)l * QKVP + 3 * HIDDEN;
  float sum = 0.f, sq = 0.f;
#pragma unroll
  for (int i = 0; i < 16; i++) { float v = seg[tid + i * 256]; sum += v; sq += v * v; }
  s1[tid] = sum; s2[tid] = sq; __syncthreads();
  for (int off = 128; off > 0; off >>= 1) {
    if (tid < off) { s1[tid] += s1[tid + off]; s2[tid] += s2[tid + off]; }
    __syncthreads();
  }
  float mean = s1[0] * (1.f / 4096.f);
  float var  = s2[0] * (1.f / 4096.f) - mean * mean;
  float rinv = rsqrtf(var + EPSV);
#pragma unroll
  for (int i = 0; i < 16; i++) {
    int c = tid + i * 256;
    float y = (seg[c] - mean) * rinv * g[c] + b[c];
    cat[(size_t)l * VP + HIDDEN + c] = (__bf16)(y > 0.f ? y : 0.f);
  }
}

// ------------------------------------------------------ bias + out_ln -> out
__global__ __launch_bounds__(256) void final_ln_kernel(const float* __restrict__ tmp,
                                                       const float* __restrict__ bias,
                                                       const float* __restrict__ g,
                                                       const float* __restrict__ b,
                                                       float* __restrict__ out) {
  __shared__ float s1[256], s2[256];
  const int l = blockIdx.x, tid = threadIdx.x;
  const float* row = tmp + (size_t)l * HIDDEN;
  float y[4]; float sum = 0.f, sq = 0.f;
#pragma unroll
  for (int i = 0; i < 4; i++) {
    int c = tid + i * 256;
    y[i] = row[c] + bias[c];
    sum += y[i]; sq += y[i] * y[i];
  }
  s1[tid] = sum; s2[tid] = sq; __syncthreads();
  for (int off = 128; off > 0; off >>= 1) {
    if (tid < off) { s1[tid] += s1[tid + off]; s2[tid] += s2[tid + off]; }
    __syncthreads();
  }
  float mean = s1[0] * (1.f / HIDDEN);
  float var  = s2[0] * (1.f / HIDDEN) - mean * mean;
  float rinv = rsqrtf(var + EPSV);
#pragma unroll
  for (int i = 0; i < 4; i++) {
    int c = tid + i * 256;
    out[(size_t)l * HIDDEN + c] = (y[i] - mean) * rinv * g[c] + b[c];
  }
}

extern "C" void kernel_launch(void* const* d_in, const int* in_sizes, int n_in,
                              void* d_out, int out_size, void* d_ws, size_t ws_size,
                              hipStream_t stream) {
  const float* x   = (const float*)d_in[0];
  const float* AL  = (const float*)d_in[1];
  const float* w1  = (const float*)d_in[2];
  const float* w2  = (const float*)d_in[3];
  const float* opb = (const float*)d_in[4];
  const float* ing = (const float*)d_in[5];  const float* inb = (const float*)d_in[6];
  const float* qg  = (const float*)d_in[7];  const float* qb  = (const float*)d_in[8];
  const float* kg  = (const float*)d_in[9];  const float* kbv = (const float*)d_in[10];
  const float* mg  = (const float*)d_in[11]; const float* mb  = (const float*)d_in[12];
  const float* og  = (const float*)d_in[13]; const float* ob  = (const float*)d_in[14];

  char* ws = (char*)d_ws;
  __bf16* h_bf  = (__bf16*)ws; ws += (size_t)SEQ * HIDDEN * 2;
  __bf16* w1_bf = (__bf16*)ws; ws += (size_t)QKVP * HIDDEN * 2;
  __bf16* w2_bf = (__bf16*)ws; ws += (size_t)HIDDEN * VP * 2;
  float*  qkvp  = (float*) ws; ws += (size_t)SEQ * QKVP * 4;
  __bf16* q_bf  = (__bf16*)ws; ws += (size_t)SEQ * HIDDEN * 2;
  __bf16* k_bf  = (__bf16*)ws; ws += (size_t)SEQ * HIDDEN * 2;
  __bf16* v_bf  = (__bf16*)ws; ws += (size_t)SEQ * HIDDEN * 2;
  __bf16* cat   = (__bf16*)ws; ws += (size_t)SEQ * VP * 2;
  float*  tmpo  = (float*) ws; ws += (size_t)SEQ * HIDDEN * 4;

  float* out_main = (float*)d_out;                       // [2048,1024]
  float* out_res  = out_main + (size_t)SEQ * HIDDEN;     // [2048,2048]

  f2bf_kernel<<<(QKVP * HIDDEN) / 256, 256, 0, stream>>>(w1, w1_bf, QKVP * HIDDEN);
  f2bf_kernel<<<(HIDDEN * VP) / 256, 256, 0, stream>>>(w2, w2_bf, HIDDEN * VP);
  ln_shift_kernel<<<SEQ, 256, 0, stream>>>(x, ing, inb, h_bf);
  gemm_nt_bf16<<<dim3(QKVP / 256, SEQ / 64), 256, 0, stream>>>(h_bf, w1_bf, qkvp, SEQ, QKVP, HIDDEN, 1.0f);
  qkv_ln_kernel<<<SEQ, 256, 0, stream>>>(qkvp, qg, qb, kg, kbv, q_bf, k_bf, v_bf);
  // logit_residual = (scale/heads) * Q K^T over full hidden dim
  gemm_nt_bf16<<<dim3(SEQ / 256, SEQ / 64), 256, 0, stream>>>(q_bf, k_bf, out_res, SEQ, SEQ, HIDDEN, 0.125f / 16.0f);
  attn_kernel<<<dim3(SEQ / 16, HEADS), 32, 0, stream>>>(q_bf, k_bf, v_bf, AL, cat);
  mid_ln_kernel<<<SEQ, 256, 0, stream>>>(qkvp, mg, mb, cat);
  gemm_nt_bf16<<<dim3(HIDDEN / 256, SEQ / 64), 256, 0, stream>>>(cat, w2_bf, tmpo, SEQ, HIDDEN, VP, 1.0f);
  final_ln_kernel<<<SEQ, 256, 0, stream>>>(tmpo, opb, og, ob, out_main);
}
